// CrossAttention_64501818851365
// MI455X (gfx1250) — matrix operand order repaired and checked by reading, not compiled, not run on hardware
//
#include <hip/hip_runtime.h>
#include <hip/hip_bf16.h>

// ---------------------------------------------------------------------------
// CrossAttention for MI455X (gfx1250, wave32, WMMA + Tensor Data Mover).
// Pipeline: fp32->bf16 cvt -> Q GEMM (scale folded) -> KV GEMM (de-interleave)
//           -> flash attention (WMMA QK^T + PV, online softmax) -> Out GEMM.
// Matmuls: v_wmma_f32_16x16x32_bf16. Tile staging: TDM tensor_load_to_lds
// (D# LDS-padding gives the bank-staggered layouts), double-buffered so the
// TDM + global loads overlap WMMA compute; one barrier per tile.
// ---------------------------------------------------------------------------

typedef __bf16 v16bf __attribute__((ext_vector_type(16)));
typedef __bf16 v8bf  __attribute__((ext_vector_type(8)));
typedef __bf16 v4bf  __attribute__((ext_vector_type(4)));
typedef float  v8f   __attribute__((ext_vector_type(8)));
typedef unsigned int u32x4 __attribute__((ext_vector_type(4)));
typedef int          i32x4 __attribute__((ext_vector_type(4)));
typedef int          i32x8 __attribute__((ext_vector_type(8)));

constexpr int B_  = 2;
constexpr int LQ_ = 2048;
constexpr int LC_ = 2048;
constexpr int M_  = 1024;
constexpr int H_  = 16;
constexpr int D_  = 64;          // head dim
constexpr float SCALE_ = 0.125f; // 64^-0.5 (folded into Q projection)

#if __has_builtin(__builtin_amdgcn_tensor_load_to_lds) && \
    __has_builtin(__builtin_amdgcn_s_wait_tensorcnt)
#define USE_TDM 1
#else
#define USE_TDM 0
#endif

// Build a 16-wide bf16 fragment from two contiguous 8-element (16B) chunks.
__device__ __forceinline__ v16bf load2x8(const __bf16* p0, const __bf16* p1) {
    v8bf a = *(const v8bf*)p0;
    v8bf b = *(const v8bf*)p1;
    v16bf r;
#pragma unroll
    for (int i = 0; i < 8; ++i) { r[i] = a[i]; r[i + 8] = b[i]; }
    return r;
}

__device__ __forceinline__ v8f v8f_zero() {
    v8f z;
#pragma unroll
    for (int i = 0; i < 8; ++i) z[i] = 0.0f;
    return z;
}

#if USE_TDM
// TDM 2D tile load: global (row-major, row_stride elems of bf16) -> LDS,
// with LDS row padding (threshold 2^(pi+1) DWORDs of data, (pa+1) DWORDs pad).
// D# per CDNA5 ISA 8.3/8.4: group0 = {count=1, lds_addr, global_addr, type=2},
// group1 = {data_size=2B, pads, dims, tile dims, stride}; groups 2/3 zero (2D).
__device__ __forceinline__ void tdm_load_2d(unsigned lds_byte_off, const void* gptr,
                                            unsigned tile_d0, unsigned tile_d1,
                                            unsigned row_stride_elems,
                                            unsigned pad_en, unsigned pad_interval,
                                            unsigned pad_amount) {
    unsigned long long ga = (unsigned long long)(uintptr_t)gptr;
    u32x4 g0;
    g0[0] = 1u;
    g0[1] = lds_byte_off;
    g0[2] = (unsigned)ga;
    g0[3] = (unsigned)((ga >> 32) & 0x1FFFFFFu) | (2u << 30);
    i32x8 g1;
    g1[0] = (int)((1u << 16) | (pad_en << 20) | (pad_interval << 22) | (pad_amount << 25));
    g1[1] = (int)(tile_d0 << 16);
    g1[2] = (int)(tile_d1 << 16);
    g1[3] = (int)(tile_d0 << 16);
    g1[4] = (int)tile_d1;
    g1[5] = (int)row_stride_elems;
    g1[6] = 0;
    g1[7] = 0;
    i32x4 z4; z4[0] = z4[1] = z4[2] = z4[3] = 0;
#if __clang_major__ >= 23
    i32x8 z8;
#pragma unroll
    for (int i = 0; i < 8; ++i) z8[i] = 0;
    __builtin_amdgcn_tensor_load_to_lds(g0, g1, z4, z4, z8, 0);
#else
    __builtin_amdgcn_tensor_load_to_lds(g0, g1, z4, z4, 0);
#endif
}
__device__ __forceinline__ unsigned lds_off(const void* p) {
    return (unsigned)(uintptr_t)p;   // shared ptr -> low 32 bits = LDS offset
}
#endif

// ---------------------------------------------------------------------------
// fp32 -> bf16 conversion (vectorized x4)
// ---------------------------------------------------------------------------
__global__ __launch_bounds__(256)
void cvt_f32_to_bf16(const float* __restrict__ in, __bf16* __restrict__ out, int n4) {
    int i = blockIdx.x * blockDim.x + threadIdx.x;
    if (i < n4) {
        float4 v = ((const float4*)in)[i];
        v4bf r;
        r[0] = (__bf16)v.x; r[1] = (__bf16)v.y;
        r[2] = (__bf16)v.z; r[3] = (__bf16)v.w;
        ((v4bf*)out)[i] = r;
    }
}

// ---------------------------------------------------------------------------
// Tiled WMMA GEMM: C[M,N] = (A[M,K] @ W[K,N] + bias) * oscale
// Block: 256 threads (8 waves), tile 128(M) x 64(N), k-step 32.
// Double-buffered: TDM stages A(k+32) and registers stage W(k+32) while the
// WMMAs consume tile k; one barrier per k-step.
// MODE 0: bf16 row-major.  MODE 1: KV de-interleave epilogue.  MODE 2: fp32.
// ---------------------------------------------------------------------------
template <int MODE>
__global__ __launch_bounds__(256)
void gemm_bf16_wmma(const __bf16* __restrict__ A, const __bf16* __restrict__ W,
                    const float* __restrict__ bias, float oscale,
                    void* __restrict__ out0, void* __restrict__ out1,
                    int M, int N, int K) {
    (void)M; (void)out1;
    __shared__ __bf16 Asl[2][128][40];  // 32 data + 8 pad (TDM: 8DW/row + 4DW)
    __shared__ __bf16 Bsl[2][64][40];   // W tile transposed: [n][k]

    const int tid  = threadIdx.x;
    const int wv   = tid >> 5;
    const int lane = tid & 31;
    const int lo16 = lane & 15;
    const bool hi  = lane >= 16;
    const int kB   = hi ? 8 : 0;
    const int kBB  = hi ? 16 : 0;

    const int m0 = blockIdx.y * 128;
    const int n0 = blockIdx.x * 64;

    v8f acc[4];
#pragma unroll
    for (int nt = 0; nt < 4; ++nt) acc[nt] = v8f_zero();

    // ---- stage A tile (TDM or cooperative) into buffer `buf` for k-offset kk
    auto stage_A = [&](int buf, int kk) {
#if USE_TDM
        if (wv == 0)
            tdm_load_2d(lds_off(&Asl[buf][0][0]), A + (size_t)m0 * K + kk,
                        32, 128, (unsigned)K, 1, /*pi*/2, /*pa*/3);
#else
        for (int e = tid; e < 128 * 4; e += 256) {
            int r = e >> 2, c4 = e & 3;
            *(v8bf*)&Asl[buf][r][c4 * 8] =
                *(const v8bf*)&A[(size_t)(m0 + r) * K + kk + c4 * 8];
        }
#endif
    };

    __bf16 wreg[8];
    auto fetch_W = [&](int kk) {
#pragma unroll
        for (int i = 0; i < 8; ++i) {
            int e = tid + i * 256;
            int k = e >> 6, n = e & 63;
            wreg[i] = W[(size_t)(kk + k) * N + n0 + n];
        }
    };
    auto drain_W = [&](int buf) {
#pragma unroll
        for (int i = 0; i < 8; ++i) {
            int e = tid + i * 256;
            int k = e >> 6, n = e & 63;
            Bsl[buf][n][k] = wreg[i];
        }
    };

    // Prologue: tile 0
    stage_A(0, 0);
    fetch_W(0);
    drain_W(0);
#if USE_TDM
    if (wv == 0) __builtin_amdgcn_s_wait_tensorcnt(0);
#endif
    __syncthreads();

    int cur = 0;
    for (int kk0 = 0; kk0 < K; kk0 += 32) {
        const int nxt = cur ^ 1;
        const bool hasNext = (kk0 + 32) < K;
        if (hasNext) {
            stage_A(nxt, kk0 + 32);   // TDM overlaps compute below
            fetch_W(kk0 + 32);        // global loads overlap compute below
        }

        const int mrow = wv * 16 + lo16;
        v16bf af = load2x8(&Asl[cur][mrow][kB], &Asl[cur][mrow][16 + kB]);
#pragma unroll
        for (int nt = 0; nt < 4; ++nt) {
            const int n = nt * 16 + lo16;
            v16bf bf_ = load2x8(&Bsl[cur][n][kB], &Bsl[cur][n][16 + kB]);
            acc[nt] = __builtin_amdgcn_wmma_f32_16x16x32_bf16(
                false, af, false, bf_, (short)0, acc[nt], false, false);
        }

        if (hasNext) {
            drain_W(nxt);
#if USE_TDM
            if (wv == 0) __builtin_amdgcn_s_wait_tensorcnt(0);
#endif
        }
        __syncthreads();
        cur = nxt;
    }

    // Epilogue
#pragma unroll
    for (int nt = 0; nt < 4; ++nt) {
        const int col = n0 + nt * 16 + lo16;
        const float bv = bias[col];
#pragma unroll
        for (int j = 0; j < 8; ++j) {
            const int row = m0 + wv * 16 + (hi ? 8 : 0) + j;
            const float val = (acc[nt][j] + bv) * oscale;
            if constexpr (MODE == 0) {
                ((__bf16*)out0)[(size_t)row * N + col] = (__bf16)val;
            } else if constexpr (MODE == 2) {
                ((float*)out0)[(size_t)row * N + col] = val;
            } else {
                const int h  = col >> 7;
                const int rr = col & 127;
                const int d  = rr >> 1;
                const int c  = rr & 1;
                const int b  = row >> 11;
                const int s  = row & 2047;
                __bf16* dst = c ? (__bf16*)out1 : (__bf16*)out0;
                dst[(((size_t)(b * H_ + h)) * LC_ + s) * D_ + d] = (__bf16)val;
            }
        }
    }
}

// ---------------------------------------------------------------------------
// Flash attention: grid = B*H*(LQ/128) blocks, 256 threads (8 waves).
// Wave w owns 16 q-rows (128 per block sharing each K/V tile).
// Double-buffered: TDM stages K(i+1), registers stage V(i+1) (transposed on
// drain) while tile i is consumed by QK^T / softmax / PV WMMAs.
// ---------------------------------------------------------------------------
__global__ __launch_bounds__(256)
void attn_wmma(const __bf16* __restrict__ qb, const __bf16* __restrict__ kbuf,
               const __bf16* __restrict__ vbuf, __bf16* __restrict__ ob) {
    __shared__ __bf16 Ksl[2][64][72];    // K tile [s][d]  (TDM pad 32DW+4DW)
    __shared__ __bf16 Vtl[2][64][72];    // V tile transposed [d][s]
    __shared__ __bf16 Psl[8][16][72];    // per-wave P (16 rows x 64 s)

    const int tid  = threadIdx.x;
    const int wv   = tid >> 5;
    const int lane = tid & 31;
    const int lo16 = lane & 15;
    const bool hi  = lane >= 16;
    const int kB   = hi ? 8 : 0;
    const int kBB  = hi ? 16 : 0;
    const int rbase = hi ? 8 : 0;

    const int bh = blockIdx.x >> 4;      // b*H + h
    const int qt = blockIdx.x & 15;      // q tile index (128 rows each)
    const int b  = bh >> 4;
    const int h  = bh & 15;

    // Q A-fragments (already scaled by SCALE_ in the Q projection).
    const int qrow = qt * 128 + wv * 16 + lo16;
    const __bf16* qptr = qb + ((size_t)(b * LQ_) + qrow) * M_ + h * D_;
    v16bf qf[2];
#pragma unroll
    for (int c = 0; c < 2; ++c)
        qf[c] = load2x8(qptr + c * 32 + kB, qptr + c * 32 + 16 + kB);

    float mrow[8], lrow[8];
    v8f o[4];
#pragma unroll
    for (int j = 0; j < 8; ++j) { mrow[j] = -3.0e38f; lrow[j] = 0.0f; }
#pragma unroll
    for (int nt = 0; nt < 4; ++nt) o[nt] = v8f_zero();

    const __bf16* kg = kbuf + (size_t)bh * LC_ * D_;
    const __bf16* vg = vbuf + (size_t)bh * LC_ * D_;

    auto stage_K = [&](int buf, int sc) {
#if USE_TDM
        if (wv == 0)
            tdm_load_2d(lds_off(&Ksl[buf][0][0]), kg + (size_t)sc * D_,
                        64, 64, 64, 1, /*pi*/4, /*pa*/3);
#else
        for (int e = tid; e < 64 * 8; e += 256) {
            int s = e >> 3, c8 = e & 7;
            *(v8bf*)&Ksl[buf][s][c8 * 8] =
                *(const v8bf*)(kg + (size_t)(sc + s) * D_ + c8 * 8);
        }
#endif
    };

    v8bf vreg[2];   // each thread: 2 x 8-elem chunks of the V tile
    auto fetch_V = [&](int sc) {
#pragma unroll
        for (int i = 0; i < 2; ++i) {
            int e = tid + i * 256;
            int s = e >> 3, c8 = e & 7;
            vreg[i] = *(const v8bf*)(vg + (size_t)(sc + s) * D_ + c8 * 8);
        }
    };
    auto drain_V = [&](int buf) {   // transpose on drain: Vtl[d][s]
#pragma unroll
        for (int i = 0; i < 2; ++i) {
            int e = tid + i * 256;
            int s = e >> 3, c8 = e & 7;
#pragma unroll
            for (int q = 0; q < 8; ++q) Vtl[buf][c8 * 8 + q][s] = vreg[i][q];
        }
    };

    // Prologue: tile 0
    stage_K(0, 0);
    fetch_V(0);
    drain_V(0);
#if USE_TDM
    if (wv == 0) __builtin_amdgcn_s_wait_tensorcnt(0);
#endif
    __syncthreads();

    int cur = 0;
    for (int sc = 0; sc < LC_; sc += 64) {
        const int nxt = cur ^ 1;
        const bool hasNext = (sc + 64) < LC_;
        if (hasNext) {
            stage_K(nxt, sc + 64);     // TDM overlaps compute
            fetch_V(sc + 64);          // global loads overlap compute
            if (tid == 0 && sc + 128 < LC_) {
                __builtin_prefetch(kg + (size_t)(sc + 128) * D_, 0, 0);
                __builtin_prefetch(vg + (size_t)(sc + 128) * D_, 0, 0);
            }
        }

        // S = Q K^T for this wave's 16 rows x 64 context cols (pre-scaled).
        v8f sf[4];
#pragma unroll
        for (int nt = 0; nt < 4; ++nt) {
            sf[nt] = v8f_zero();
            const int n = nt * 16 + lo16;
#pragma unroll
            for (int c = 0; c < 2; ++c) {
                v16bf kf = load2x8(&Ksl[cur][n][c * 32 + kB],
                                   &Ksl[cur][n][c * 32 + 16 + kB]);
                sf[nt] = __builtin_amdgcn_wmma_f32_16x16x32_bf16(
                    false, qf[c], false, kf, (short)0, sf[nt], false, false);
            }
        }

        // Online softmax (C-frag: element j = row rbase+j; width-16 reductions).
#pragma unroll
        for (int j = 0; j < 8; ++j) {
            float rmax = fmaxf(fmaxf(sf[0][j], sf[1][j]), fmaxf(sf[2][j], sf[3][j]));
#pragma unroll
            for (int off = 8; off >= 1; off >>= 1)
                rmax = fmaxf(rmax, __shfl_xor(rmax, off, 16));
            const float mnew = fmaxf(mrow[j], rmax);
            const float alpha = __expf(mrow[j] - mnew);
            mrow[j] = mnew;
            float rsum = 0.0f;
#pragma unroll
            for (int nt = 0; nt < 4; ++nt) {
                float p = __expf(sf[nt][j] - mnew);
                sf[nt][j] = p;
                rsum += p;
            }
#pragma unroll
            for (int off = 8; off >= 1; off >>= 1)
                rsum += __shfl_xor(rsum, off, 16);
            lrow[j] = lrow[j] * alpha + rsum;
#pragma unroll
            for (int nt = 0; nt < 4; ++nt) o[nt][j] *= alpha;
        }

        // P: C-layout -> per-wave LDS -> A-layout fragments.
#pragma unroll
        for (int nt = 0; nt < 4; ++nt)
#pragma unroll
            for (int j = 0; j < 8; ++j)
                Psl[wv][rbase + j][nt * 16 + lo16] = (__bf16)sf[nt][j];
        asm volatile("" ::: "memory");  // same-wave LDS ops are in-order

        // O += P @ V
#pragma unroll
        for (int c = 0; c < 2; ++c) {
            v16bf pa = load2x8(&Psl[wv][lo16][c * 32 + kB],
                               &Psl[wv][lo16][c * 32 + 16 + kB]);
#pragma unroll
            for (int nt = 0; nt < 4; ++nt) {
                const int d = nt * 16 + lo16;
                v16bf vf = load2x8(&Vtl[cur][d][c * 32 + kB],
                                   &Vtl[cur][d][c * 32 + 16 + kB]);
                o[nt] = __builtin_amdgcn_wmma_f32_16x16x32_bf16(
                    false, pa, false, vf, (short)0, o[nt], false, false);
            }
        }
        asm volatile("" ::: "memory");

        if (hasNext) {
            drain_V(nxt);
#if USE_TDM
            if (wv == 0) __builtin_amdgcn_s_wait_tensorcnt(0);
#endif
        }
        __syncthreads();
        cur = nxt;
    }

    // Normalize and write [B, LQ, H*D] bf16.
#pragma unroll
    for (int j = 0; j < 8; ++j) {
        const float rinv = 1.0f / lrow[j];
        const int qr = qt * 128 + wv * 16 + rbase + j;
        const size_t base = ((size_t)(b * LQ_) + qr) * M_ + h * D_;
#pragma unroll
        for (int nt = 0; nt < 4; ++nt)
            ob[base + nt * 16 + lo16] = (__bf16)(o[nt][j] * rinv);
    }
}

// ---------------------------------------------------------------------------
// Host launcher
// ---------------------------------------------------------------------------
extern "C" void kernel_launch(void* const* d_in, const int* in_sizes, int n_in,
                              void* d_out, int out_size, void* d_ws, size_t ws_size,
                              hipStream_t stream) {
    (void)in_sizes; (void)n_in; (void)out_size; (void)ws_size;

    const float* query   = (const float*)d_in[0];
    const float* context = (const float*)d_in[1];
    const float* Wq      = (const float*)d_in[2];
    const float* bq      = (const float*)d_in[3];
    const float* Wkv     = (const float*)d_in[4];
    const float* bkv     = (const float*)d_in[5];
    const float* Wout    = (const float*)d_in[6];
    const float* bout    = (const float*)d_in[7];
    float* out = (float*)d_out;

    char* ws = (char*)d_ws;
    size_t off = 0;
    auto carve = [&](size_t elems) { void* p = ws + off; off += elems * 2; return p; };
    const size_t ACT = (size_t)B_ * LQ_ * M_;
    __bf16* query_bf   = (__bf16*)carve(ACT);
    __bf16* context_bf = (__bf16*)carve(ACT);
    __bf16* Wq_bf      = (__bf16*)carve((size_t)M_ * M_);
    __bf16* Wkv_bf     = (__bf16*)carve((size_t)M_ * 2 * M_);
    __bf16* Wout_bf    = (__bf16*)carve((size_t)M_ * M_);
    __bf16* qb      = (__bf16*)carve(ACT);
    __bf16* kbuf    = (__bf16*)carve((size_t)B_ * H_ * LC_ * D_);
    __bf16* vbuf    = (__bf16*)carve((size_t)B_ * H_ * LC_ * D_);
    __bf16* attn_bf = (__bf16*)carve(ACT);

    auto cvt = [&](const float* src, __bf16* dst, size_t n) {
        int n4 = (int)(n / 4);
        cvt_f32_to_bf16<<<(n4 + 255) / 256, 256, 0, stream>>>(src, dst, n4);
    };
    cvt(query,   query_bf,   ACT);
    cvt(context, context_bf, ACT);
    cvt(Wq,   Wq_bf,   (size_t)M_ * M_);
    cvt(Wkv,  Wkv_bf,  (size_t)M_ * 2 * M_);
    cvt(Wout, Wout_bf, (size_t)M_ * M_);

    const int ROWS = B_ * LQ_;  // 4096
    gemm_bf16_wmma<0><<<dim3(M_ / 64, ROWS / 128), 256, 0, stream>>>(
        query_bf, Wq_bf, bq, SCALE_, qb, nullptr, ROWS, M_, M_);
    gemm_bf16_wmma<1><<<dim3(2 * M_ / 64, ROWS / 128), 256, 0, stream>>>(
        context_bf, Wkv_bf, bkv, 1.0f, kbuf, vbuf, ROWS, 2 * M_, M_);
    attn_wmma<<<B_ * H_ * (LQ_ / 128), 256, 0, stream>>>(qb, kbuf, vbuf, attn_bf);
    gemm_bf16_wmma<2><<<dim3(M_ / 64, ROWS / 128), 256, 0, stream>>>(
        attn_bf, Wout_bf, bout, 1.0f, out, nullptr, ROWS, M_, M_);
}
